// CDCK2_11081015624180
// MI455X (gfx1250) — compile-verified, matrix operand order checked by reading, not run on hardware
//
#include <hip/hip_runtime.h>
#include <hip/hip_bf16.h>
#include <math.h>

typedef __attribute__((ext_vector_type(16))) __bf16 v16bf;
typedef __attribute__((ext_vector_type(8)))  float  v8f;

__device__ __forceinline__ __bf16 f2bf(float f) {
  union { float f; unsigned u; } a; a.f = f;
  unsigned r = a.u + 0x7FFFu + ((a.u >> 16) & 1u);
  unsigned short s = (unsigned short)(r >> 16);
  __bf16 b;
  __builtin_memcpy(&b, &s, 2);
  return b;
}

__device__ __forceinline__ float sigm(float x) { return 1.0f / (1.0f + expf(-x)); }

// ---------------------------------------------------------------------------
// Conv layer 1: Cin=1, k=10, s=5, p=3, Lin=20480 -> Lout=4096  (direct, mem-bound)
// ---------------------------------------------------------------------------
__global__ void __launch_bounds__(256) conv1_direct(const float* __restrict__ X,
                                                    const float* __restrict__ W,
                                                    float* __restrict__ Y) {
  size_t i = (size_t)blockIdx.x * 256 + threadIdx.x;   // i = ((b*512)+co)*4096+lo
  int lo = (int)(i & 4095);
  int co = (int)((i >> 12) & 511);
  int b  = (int)(i >> 21);
  float s = 0.f;
#pragma unroll
  for (int kt = 0; kt < 10; ++kt) {
    int pos = lo * 5 - 3 + kt;
    if (pos >= 0 && pos < 20480) s += W[co * 10 + kt] * X[(size_t)b * 20480 + pos];
  }
  Y[i] = s;
}

// ---------------------------------------------------------------------------
// BatchNorm: per-channel stats (mean, rstd), then scale/shift + ReLU in place
// ---------------------------------------------------------------------------
__global__ void __launch_bounds__(256) bn_stats(const float* __restrict__ Y, int L,
                                                float* __restrict__ stats) {
  __shared__ float s1[256], s2[256];
  int c = blockIdx.x;
  float sum = 0.f, sq = 0.f;
  for (int b = 0; b < 64; ++b) {
    const float* p = Y + ((size_t)b * 512 + c) * L;
    for (int l = threadIdx.x; l < L; l += 256) { float v = p[l]; sum += v; sq += v * v; }
  }
  s1[threadIdx.x] = sum; s2[threadIdx.x] = sq;
  __syncthreads();
  for (int off = 128; off > 0; off >>= 1) {
    if (threadIdx.x < off) { s1[threadIdx.x] += s1[threadIdx.x + off]; s2[threadIdx.x] += s2[threadIdx.x + off]; }
    __syncthreads();
  }
  if (threadIdx.x == 0) {
    float inv  = 1.0f / (64.0f * (float)L);
    float mean = s1[0] * inv;
    float var  = s2[0] * inv - mean * mean;
    stats[c * 2]     = mean;
    stats[c * 2 + 1] = rsqrtf(var + 1e-5f);
  }
}

__global__ void __launch_bounds__(256) bn_apply(float* __restrict__ Y, int L,
                                                const float* __restrict__ stats,
                                                const float* __restrict__ gamma,
                                                const float* __restrict__ beta) {
  size_t i = (size_t)blockIdx.x * 256 + threadIdx.x;
  int c = (int)((i / (size_t)L) % 512);
  float m = stats[2 * c], r = stats[2 * c + 1];
  float v = (Y[i] - m) * r * gamma[c] + beta[c];
  Y[i] = v > 0.f ? v : 0.f;
}

// ---------------------------------------------------------------------------
// Implicit-GEMM conv via WMMA bf16 (Cout=512).  Block: 64(Cout) x 64(pos),
// 256 threads = 8 waves, each wave: one 16-row tile x two 16-col tiles.
// Software-pipelined: chunk k+1's global loads are staged in VGPRs while the
// WMMAs for chunk k run; convert+swizzle into LDS happens behind the barrier.
// LDS fragment layout (16-bit A/B): lane=(r&15)+16*((k>>3)&1), j=(k&7)+8*(k>=16)
// ---------------------------------------------------------------------------
__global__ void __launch_bounds__(256) conv_wmma_gemm(
    const float* __restrict__ X, const float* __restrict__ W, float* __restrict__ Y,
    int Cin, int Lin, int Lout, int ksize, int stride, int pad) {
  __shared__ alignas(32) __bf16 ldsA[4][32][16];
  __shared__ alignas(32) __bf16 ldsB[4][32][16];

  const int tid  = threadIdx.x;
  const int lane = tid & 31;
  const int wv   = tid >> 5;
  const int rt   = wv & 3;
  const int ct0  = (wv >> 2) * 2;
  const int loB  = blockIdx.x * 64;
  const int coB  = blockIdx.y * 64;
  const int bz   = blockIdx.z;
  const int K    = Cin * ksize;

  // loop-invariant load decomposition (elem = tid + i*256)
  const int kkA   = tid & 31;                       // A-side k within chunk
  const int khA   = (kkA >> 3) & 1;
  const int jA    = (kkA & 7) + ((kkA >> 4) << 3);
  const int rowA0 = tid >> 5;                       // A row for i=0 (stride 8)
  const int colB  = tid & 63;                       // B-side column
  const int ctB   = colB >> 4;
  const int lnB   = colB & 15;
  const int nB_   = loB + colB;
  const int pos0  = nB_ * stride - pad;
  const bool colOK = (nB_ < Lout);
  const size_t wrowstride = (size_t)Cin * ksize;

  float ra[8], rb[8];
  auto loadChunk = [&](int kb) {
    int kgA = kb + kkA;
    int ciA = kgA / ksize, ktA = kgA - ciA * ksize;
    const float* WA = W + ((size_t)coB * Cin + ciA) * ksize + ktA;
#pragma unroll
    for (int i = 0; i < 8; ++i) ra[i] = WA[(size_t)(rowA0 + i * 8) * wrowstride];
#pragma unroll
    for (int i = 0; i < 8; ++i) {
      int kk = (tid >> 6) + i * 4;
      int kg = kb + kk;
      int ci = kg / ksize, kt = kg - ci * ksize;
      int pos = pos0 + kt;
      rb[i] = (colOK && pos >= 0 && pos < Lin)
                  ? X[((size_t)bz * Cin + ci) * Lin + pos] : 0.0f;
    }
  };
  auto storeChunk = [&]() {
#pragma unroll
    for (int i = 0; i < 8; ++i) {
      int row = rowA0 + i * 8;
      ldsA[row >> 4][(row & 15) + (khA << 4)][jA] = f2bf(ra[i]);
    }
#pragma unroll
    for (int i = 0; i < 8; ++i) {
      int kk = (tid >> 6) + i * 4;
      int kh = (kk >> 3) & 1, j = (kk & 7) + ((kk >> 4) << 3);
      ldsB[ctB][lnB + (kh << 4)][j] = f2bf(rb[i]);
    }
  };

  v8f acc0 = {};
  v8f acc1 = {};

  loadChunk(0);
  for (int kb = 0; kb < K; kb += 32) {
    __syncthreads();                 // previous WMMAs done reading LDS
    storeChunk();
    __syncthreads();
    if (kb + 32 < K) {
      loadChunk(kb + 32);            // overlaps with the WMMAs below
      int kg2 = kb + 32 + kkA;
      int ci2 = kg2 / ksize, kt2 = kg2 - ci2 * ksize;
      __builtin_prefetch(W + ((size_t)(coB + rowA0) * Cin + ci2) * ksize + kt2, 0, 0);
    }
    v16bf a  = *(const v16bf*)(&ldsA[rt][lane][0]);
    v16bf b0 = *(const v16bf*)(&ldsB[ct0][lane][0]);
    v16bf b1 = *(const v16bf*)(&ldsB[ct0 + 1][lane][0]);
    acc0 = __builtin_amdgcn_wmma_f32_16x16x32_bf16(false, a, false, b0, (short)0, acc0, false, false);
    acc1 = __builtin_amdgcn_wmma_f32_16x16x32_bf16(false, a, false, b1, (short)0, acc1, false, false);
  }
#pragma unroll
  for (int vi = 0; vi < 8; ++vi) {
    int co = coB + rt * 16 + vi + ((lane >> 4) << 3);
    int n0 = loB + ct0 * 16 + (lane & 15);
    int n1 = n0 + 16;
    if (n0 < Lout) Y[((size_t)bz * 512 + co) * Lout + n0] = acc0[vi];
    if (n1 < Lout) Y[((size_t)bz * 512 + co) * Lout + n1] = acc1[vi];
  }
}

// ---------------------------------------------------------------------------
// x_proj GEMM: out[(b,t),g] = sum_d wih[g,d]*Z[b,d,t] + bih[g]
//   M=768 (grid.y=12), N=12864=201*64 (grid.x=201), K=512 (same pipelining)
// ---------------------------------------------------------------------------
__global__ void __launch_bounds__(256) gemm_xproj(
    const float* __restrict__ Z, const float* __restrict__ Wih,
    const float* __restrict__ bih, float* __restrict__ XP) {
  __shared__ alignas(32) __bf16 ldsA[4][32][16];
  __shared__ alignas(32) __bf16 ldsB[4][32][16];
  const int tid  = threadIdx.x;
  const int lane = tid & 31;
  const int wv   = tid >> 5;
  const int rt   = wv & 3;
  const int ct0  = (wv >> 2) * 2;
  const int nB   = blockIdx.x * 64;
  const int coB  = blockIdx.y * 64;

  const int kkA   = tid & 31;
  const int khA   = (kkA >> 3) & 1;
  const int jA    = (kkA & 7) + ((kkA >> 4) << 3);
  const int rowA0 = tid >> 5;
  const int colB  = tid & 63;
  const int ctB   = colB >> 4;
  const int lnB   = colB & 15;
  const int nCol  = nB + colB;
  const int bbB   = nCol / 201, ttB = nCol - bbB * 201;
  const float* Zcol = Z + (size_t)bbB * 512 * 409 + ttB;

  float ra[8], rb[8];
  auto loadChunk = [&](int kb) {
#pragma unroll
    for (int i = 0; i < 8; ++i)
      ra[i] = Wih[(size_t)(coB + rowA0 + i * 8) * 512 + kb + kkA];
#pragma unroll
    for (int i = 0; i < 8; ++i) {
      int kk = (tid >> 6) + i * 4;
      rb[i] = Zcol[(size_t)(kb + kk) * 409];
    }
  };
  auto storeChunk = [&]() {
#pragma unroll
    for (int i = 0; i < 8; ++i) {
      int row = rowA0 + i * 8;
      ldsA[row >> 4][(row & 15) + (khA << 4)][jA] = f2bf(ra[i]);
    }
#pragma unroll
    for (int i = 0; i < 8; ++i) {
      int kk = (tid >> 6) + i * 4;
      int kh = (kk >> 3) & 1, j = (kk & 7) + ((kk >> 4) << 3);
      ldsB[ctB][lnB + (kh << 4)][j] = f2bf(rb[i]);
    }
  };

  v8f acc0 = {}, acc1 = {};
  loadChunk(0);
  for (int kb = 0; kb < 512; kb += 32) {
    __syncthreads();
    storeChunk();
    __syncthreads();
    if (kb + 32 < 512) loadChunk(kb + 32);
    v16bf a  = *(const v16bf*)(&ldsA[rt][lane][0]);
    v16bf b0 = *(const v16bf*)(&ldsB[ct0][lane][0]);
    v16bf b1 = *(const v16bf*)(&ldsB[ct0 + 1][lane][0]);
    acc0 = __builtin_amdgcn_wmma_f32_16x16x32_bf16(false, a, false, b0, (short)0, acc0, false, false);
    acc1 = __builtin_amdgcn_wmma_f32_16x16x32_bf16(false, a, false, b1, (short)0, acc1, false, false);
  }
#pragma unroll
  for (int vi = 0; vi < 8; ++vi) {
    int co = coB + rt * 16 + vi + ((lane >> 4) << 3);
    int n0 = nB + ct0 * 16 + (lane & 15);
    int n1 = n0 + 16;
    {
      int bb = n0 / 201, tt = n0 - bb * 201;
      XP[((size_t)bb * 201 + tt) * 768 + co] = acc0[vi] + bih[co];
    }
    {
      int bb = n1 / 201, tt = n1 - bb * 201;
      XP[((size_t)bb * 201 + tt) * 768 + co] = acc1[vi] + bih[co];
    }
  }
}

// ---------------------------------------------------------------------------
// One-time pre-swizzle of whh (768x256 f32) into bf16 WMMA A-fragment layout:
//   Wswz[rtile(48)][kc(8)][lane(32)][j(16)]
// ---------------------------------------------------------------------------
__global__ void __launch_bounds__(256) whh_prep(const float* __restrict__ Whh,
                                                __bf16* __restrict__ Wswz) {
  int i = blockIdx.x * 256 + threadIdx.x;    // 768*256
  int g = i >> 8, d = i & 255;
  int rtile = g >> 4, r = g & 15;
  int kc = d >> 5, kk = d & 31;
  int kh = (kk >> 3) & 1;
  int j  = (kk & 7) + ((kk >> 4) << 3);
  Wswz[(((size_t)rtile * 8 + kc) * 32 + (r + (kh << 4))) * 16 + j] =
      f2bf(Whh[(size_t)g * 256 + d]);
}

// ---------------------------------------------------------------------------
// Persistent GRU: the whole 201-step recurrence in ONE workgroup (16 waves),
// exploiting CDNA5's 320KB LDS/WGP.  Dynamic LDS (288KB):
//   hF   : f32  h state            64*256*4  =  64KB
//   hp   : f32  whh @ h + bhh      64*768*4  = 192KB
//   hswz : bf16 h in B-frag layout 4*8*32*16 =  32KB
// Each wave owns 3 row-tiles x 4 col-tiles; A-fragments (whh) are loaded once
// per row-tile into registers and reused across all 4 col-tiles -> 4x less L2
// traffic on the serialized critical path (384KB/step instead of 1536KB/step).
// ---------------------------------------------------------------------------
#define GRU_THREADS 512
__global__ void __launch_bounds__(GRU_THREADS) gru_persistent(
    const float* __restrict__ XP,      // [64][201][768]
    const __bf16* __restrict__ Wswz,   // [48][8][32][16]
    const float* __restrict__ bhh,     // [768]
    const float* __restrict__ hidden,  // [64][256] initial h
    float* __restrict__ Hout) {        // [64][256] final h
  extern __shared__ char smem[];
  float*  hF   = (float*)smem;                         // 65536 B
  float*  hp   = (float*)(smem + 65536);               // 196608 B
  __bf16* hswz = (__bf16*)(smem + 65536 + 196608);     // 32768 B

  const int tid  = threadIdx.x;
  const int lane = tid & 31;
  const int wv   = tid >> 5;                           // 0..15

  // init h (f32 + swizzled bf16 copy)
  for (int i = tid; i < 64 * 256; i += GRU_THREADS) {
    float v = hidden[i];
    hF[i] = v;
    int b = i >> 8, d = i & 255;
    int kc = d >> 5, kk = d & 31;
    int kh = (kk >> 3) & 1;
    int j  = (kk & 7) + ((kk >> 4) << 3);
    hswz[((((b >> 4) * 8 + kc) * 32) + ((b & 15) + (kh << 4))) * 16 + j] = f2bf(v);
  }
  __syncthreads();

  for (int t = 0; t <= 200; ++t) {
    // hp[b][g] = sum_d whh[g][d]*h[b][d] + bhh[g]
    for (int rq = 0; rq < 3; ++rq) {
      int rt = wv * 3 + rq;                       // row tile 0..47
      v16bf afr[8];
#pragma unroll
      for (int kc = 0; kc < 8; ++kc)
        afr[kc] = *(const v16bf*)(Wswz + (((size_t)rt * 8 + kc) * 32 + lane) * 16);
#pragma unroll
      for (int ct = 0; ct < 4; ++ct) {            // col tile (batch group)
        v8f acc = {};
#pragma unroll
        for (int kc = 0; kc < 8; ++kc) {
          v16bf bf = *(const v16bf*)(hswz + (((size_t)ct * 8 + kc) * 32 + lane) * 16);
          acc = __builtin_amdgcn_wmma_f32_16x16x32_bf16(false, afr[kc], false, bf,
                                                        (short)0, acc, false, false);
        }
#pragma unroll
        for (int vi = 0; vi < 8; ++vi) {
          int g = rt * 16 + vi + ((lane >> 4) << 3);
          int b = ct * 16 + (lane & 15);
          hp[b * 768 + g] = acc[vi] + bhh[g];
        }
      }
    }
    __syncthreads();
    // gates + h update (+ rebuild bf16 swizzled h)
    for (int i = tid; i < 64 * 256; i += GRU_THREADS) {
      int b = i >> 8, d = i & 255;
      size_t xb = ((size_t)b * 201 + t) * 768;
      float r  = sigm(XP[xb + d]        + hp[b * 768 + d]);
      float zg = sigm(XP[xb + 256 + d]  + hp[b * 768 + 256 + d]);
      float n  = tanhf(XP[xb + 512 + d] + r * hp[b * 768 + 512 + d]);
      float hn = (1.0f - zg) * n + zg * hF[i];
      hF[i] = hn;
      int kc = d >> 5, kk = d & 31;
      int kh = (kk >> 3) & 1;
      int j  = (kk & 7) + ((kk >> 4) << 3);
      hswz[((((b >> 4) * 8 + kc) * 32) + ((b & 15) + (kh << 4))) * 16 + j] = f2bf(hn);
    }
    __syncthreads();
  }
  for (int i = tid; i < 64 * 256; i += GRU_THREADS) Hout[i] = hF[i];
}

// ---------------------------------------------------------------------------
// Epilogue kernels (tiny)
// ---------------------------------------------------------------------------
__global__ void __launch_bounds__(256) copy_h(const float* __restrict__ H,
                                              float* __restrict__ out) {
  int i = blockIdx.x * 256 + threadIdx.x;
  out[2 + i] = H[i];
}

// pred[(t*64+b)*512+k] = sum_d h[b,d]*wkw[t,k,d] + wkb[t,k]
__global__ void __launch_bounds__(256) pred_kernel(const float* __restrict__ H,
                                                   const float* __restrict__ wkw,
                                                   const float* __restrict__ wkb,
                                                   float* __restrict__ PRED) {
  int i = blockIdx.x * 256 + threadIdx.x;      // 12*64*512
  int t = i >> 15;
  int r = i & 32767;
  int b = r >> 9, k = r & 511;
  const float* hv = H + (size_t)b * 256;
  const float* wv = wkw + ((size_t)t * 512 + k) * 256;
  float s = 0.f;
  for (int d = 0; d < 256; ++d) s += hv[d] * wv[d];
  PRED[((size_t)t * 64 + b) * 512 + k] = s + wkb[t * 512 + k];
}

// total[t,b,c] = sum_k Z[b,k,201+t] * pred[t,c,k]
__global__ void __launch_bounds__(256) total_kernel(const float* __restrict__ Z,
                                                    const float* __restrict__ PRED,
                                                    float* __restrict__ TOT) {
  int i = blockIdx.x * 256 + threadIdx.x;      // 12*64*64
  int t = i >> 12;
  int b = (i >> 6) & 63;
  int c = i & 63;
  const float* pv = PRED + ((size_t)t * 64 + c) * 512;
  float s = 0.f;
  for (int k = 0; k < 512; ++k)
    s += Z[((size_t)b * 512 + k) * 409 + 201 + t] * pv[k];
  TOT[((size_t)t * 64 + b) * 64 + c] = s;
}

// per-t partial: sum_b (total[t,b,b] - logsumexp_c total[t,b,c])
__global__ void __launch_bounds__(64) nce_kernel(const float* __restrict__ TOT,
                                                 float* __restrict__ partial) {
  __shared__ float red[64];
  int t = blockIdx.x, b = threadIdx.x;
  const float* row = TOT + ((size_t)t * 64 + b) * 64;
  float m = -1e30f;
  for (int c = 0; c < 64; ++c) m = fmaxf(m, row[c]);
  float s = 0.f;
  for (int c = 0; c < 64; ++c) s += expf(row[c] - m);
  red[b] = row[b] - (logf(s) + m);
  __syncthreads();
  for (int off = 32; off > 0; off >>= 1) {
    if (b < off) red[b] += red[b + off];
    __syncthreads();
  }
  if (b == 0) partial[t] = red[0];
}

// accuracy from total[11] (+ finalize nce)
__global__ void __launch_bounds__(64) acc_final(const float* __restrict__ TOT,
                                                const float* __restrict__ partial,
                                                float* __restrict__ out) {
  __shared__ float lse[64];
  __shared__ float cnt[64];
  int tid = threadIdx.x;
  const float* row = TOT + (size_t)(11 * 64 + tid) * 64;
  float m = -1e30f;
  for (int c = 0; c < 64; ++c) m = fmaxf(m, row[c]);
  float s = 0.f;
  for (int c = 0; c < 64; ++c) s += expf(row[c] - m);
  lse[tid] = logf(s) + m;
  __syncthreads();
  float best = -1e30f; int bi = 0;
  for (int b2 = 0; b2 < 64; ++b2) {
    float v = TOT[(size_t)(11 * 64 + b2) * 64 + tid] - lse[b2];
    if (v > best) { best = v; bi = b2; }
  }
  cnt[tid] = (bi == tid) ? 1.f : 0.f;
  __syncthreads();
  for (int off = 32; off > 0; off >>= 1) {
    if (tid < off) cnt[tid] += cnt[tid + off];
    __syncthreads();
  }
  if (tid == 0) {
    out[0] = cnt[0] * (1.0f / 64.0f);
    float ns = 0.f;
    for (int t2 = 0; t2 < 12; ++t2) ns += partial[t2];
    out[1] = ns * (-1.0f / 768.0f);
  }
}

// ---------------------------------------------------------------------------
// Host launcher
// ---------------------------------------------------------------------------
extern "C" void kernel_launch(void* const* d_in, const int* in_sizes, int n_in,
                              void* d_out, int out_size, void* d_ws, size_t ws_size,
                              hipStream_t stream) {
  const float* x      = (const float*)d_in[0];
  const float* hidden = (const float*)d_in[1];
  const float* gamma  = (const float*)d_in[2];
  const float* beta   = (const float*)d_in[3];
  const float* wih    = (const float*)d_in[4];
  const float* whh    = (const float*)d_in[5];
  const float* bih    = (const float*)d_in[6];
  const float* bhh    = (const float*)d_in[7];
  const float* wkw    = (const float*)d_in[8];
  const float* wkb    = (const float*)d_in[9];
  // d_in[10] = t_samples (hardcoded 200 per setup)
  const float* w1 = (const float*)d_in[11];
  const float* w2 = (const float*)d_in[12];
  const float* w3 = (const float*)d_in[13];
  const float* w4 = (const float*)d_in[14];
  const float* w5 = (const float*)d_in[15];
  float* out = (float*)d_out;
  char* ws = (char*)d_ws;

  constexpr size_t OFF_Y1 = 0;                    // 64*512*4096*4 = 512MB
  constexpr size_t OFF_Y3 = 0;                    // reuse (Y1 dead)
  constexpr size_t OFF_Y5 = 67108864ull;          // 64 MiB
  constexpr size_t OFF_XP = 134217728ull;         // 128 MiB
  constexpr size_t OFF_Y2 = 536870912ull;         // after Y1
  constexpr size_t OFF_Y4 = OFF_Y2;               // reuse (Y2 dead)
  constexpr size_t OFF_SM = OFF_Y2 + 107347968ull;
  float*  Y1   = (float*)(ws + OFF_Y1);
  float*  Y2   = (float*)(ws + OFF_Y2);
  float*  Y3   = (float*)(ws + OFF_Y3);
  float*  Y4   = (float*)(ws + OFF_Y4);
  float*  Y5   = (float*)(ws + OFF_Y5);
  float*  XP   = (float*)(ws + OFF_XP);
  float*  STAT = (float*)(ws + OFF_SM);                   // 512*2 f32
  float*  H    = (float*)(ws + OFF_SM + 4096);            // 64*256 f32
  float*  PRED = (float*)(ws + OFF_SM + 69632);           // 12*64*512 f32
  float*  TOT  = (float*)(ws + OFF_SM + 1642496);         // 12*64*64 f32
  float*  PART = (float*)(ws + OFF_SM + 1839104);         // 12 f32
  __bf16* WSWZ = (__bf16*)(ws + OFF_SM + 1839168);        // 768*256 bf16

  // ---- encoder ----
  conv1_direct<<<dim3(524288), 256, 0, stream>>>(x, w1, Y1);
  bn_stats<<<dim3(512), 256, 0, stream>>>(Y1, 4096, STAT);
  bn_apply<<<dim3(524288), 256, 0, stream>>>(Y1, 4096, STAT, gamma, beta);

  conv_wmma_gemm<<<dim3(13, 8, 64), 256, 0, stream>>>(Y1, w2, Y2, 512, 4096, 819, 8, 5, 2);
  bn_stats<<<dim3(512), 256, 0, stream>>>(Y2, 819, STAT);
  bn_apply<<<dim3(104832), 256, 0, stream>>>(Y2, 819, STAT, gamma + 512, beta + 512);

  conv_wmma_gemm<<<dim3(7, 8, 64), 256, 0, stream>>>(Y2, w3, Y3, 512, 819, 409, 4, 2, 1);
  bn_stats<<<dim3(512), 256, 0, stream>>>(Y3, 409, STAT);
  bn_apply<<<dim3(52352), 256, 0, stream>>>(Y3, 409, STAT, gamma + 1024, beta + 1024);

  conv_wmma_gemm<<<dim3(7, 8, 64), 256, 0, stream>>>(Y3, w4, Y4, 512, 409, 409, 3, 1, 1);
  bn_stats<<<dim3(512), 256, 0, stream>>>(Y4, 409, STAT);
  bn_apply<<<dim3(52352), 256, 0, stream>>>(Y4, 409, STAT, gamma + 1536, beta + 1536);

  conv_wmma_gemm<<<dim3(7, 8, 64), 256, 0, stream>>>(Y4, w5, Y5, 512, 409, 409, 3, 1, 1);
  bn_stats<<<dim3(512), 256, 0, stream>>>(Y5, 409, STAT);
  bn_apply<<<dim3(52352), 256, 0, stream>>>(Y5, 409, STAT, gamma + 2048, beta + 2048);

  // ---- GRU (x_proj GEMM + single persistent kernel for the recurrence) ----
  gemm_xproj<<<dim3(201, 12), 256, 0, stream>>>(Y5, wih, bih, XP);
  whh_prep<<<dim3(768), 256, 0, stream>>>(whh, WSWZ);
  gru_persistent<<<dim3(1), GRU_THREADS, 294912, stream>>>(XP, WSWZ, bhh, hidden, H);
  copy_h<<<dim3(64), 256, 0, stream>>>(H, out);   // h_last (== c_t)

  // ---- CPC scores / NCE / accuracy ----
  pred_kernel<<<dim3(1536), 256, 0, stream>>>(H, wkw, wkb, PRED);
  total_kernel<<<dim3(192), 256, 0, stream>>>(Y5, PRED, TOT);
  nce_kernel<<<dim3(12), 64, 0, stream>>>(TOT, PART);
  acc_final<<<dim3(1), 64, 0, stream>>>(TOT, PART, out);
}
